// CascadeRoIHeads_20263655702897
// MI455X (gfx1250) — matrix operand order repaired and checked by reading, not compiled, not run on hardware
//
#include <hip/hip_runtime.h>
#include <hip/hip_bf16.h>
#include <math.h>

typedef __bf16 bf16;
typedef __attribute__((ext_vector_type(16))) __bf16 v16bf;
typedef __attribute__((ext_vector_type(8)))  float  v8f;
typedef __attribute__((ext_vector_type(4)))  unsigned int u32x4;

#define BIMG 2
#define NROI 1000
#define MTOT (BIMG * NROI)          // 2000 rows through FC stack
#define CFEAT 256
#define FH 100
#define FW 152
#define POOL 7
#define DFEAT (CFEAT * POOL * POOL) // 12544
#define HID 1024
#define NCLS 81
#define NREG (NCLS * 4)             // 324
#define IMGW 1216.0f
#define IMGH 800.0f
#define BBOX_CLIP_F 4.135166556742356f
#define NSORT 131072                // next pow2 above 80000
#define NCAND 80000                 // 1000 rois * 80 fg classes
#define PRE_NMS 2000
#define DETS 500

// GEMM tiling: block tile 128x64, BK=64, double-buffered dynamic LDS
#define GEMM_LDS_BYTES 49152        // 2*(128*64 + 64*64)*2B

static __device__ __forceinline__ int imin(int a, int b) { return a < b ? a : b; }
static __device__ __forceinline__ int imax(int a, int b) { return a > b ? a : b; }

// ---------------------------------------------------------------------------
// fp32 -> bf16 conversion (weights re-converted every call: no cached state)
// ---------------------------------------------------------------------------
__global__ void cvt_f32_bf16(const float* __restrict__ s, bf16* __restrict__ d, int n) {
    int i = blockIdx.x * 256 + threadIdx.x;
    if (i < n) d[i] = (bf16)s[i];
}

// ---------------------------------------------------------------------------
// RoI align (RATIO=2, OUT=7, SCALE=0.125) -> bf16 activations (2000 x 12544)
// feature order per roi: c*49 + oy*7 + ox (matches reference transpose/reshape)
// ---------------------------------------------------------------------------
__global__ void roi_align_kernel(const float* __restrict__ feat,
                                 const float* __restrict__ rois,
                                 bf16* __restrict__ X, int total) {
    int i = blockIdx.x * blockDim.x + threadIdx.x;
    if (i >= total) return;
    int ox = i % POOL; int t = i / POOL;
    int oy = t % POOL; t /= POOL;
    int c  = t % CFEAT; t /= CFEAT;
    int roi = t % NROI; int b = t / NROI;

    const float* r = rois + ((size_t)b * NROI + roi) * 4;
    float x1 = r[0] * 0.125f, y1 = r[1] * 0.125f;
    float rw = fmaxf(r[2] * 0.125f - x1, 1.0f);
    float rh = fmaxf(r[3] * 0.125f - y1, 1.0f);
    float bw = rw * (1.0f / POOL), bh = rh * (1.0f / POOL);
    const float* fm = feat + ((size_t)b * CFEAT + c) * (FH * FW);

    float acc = 0.0f;
#pragma unroll
    for (int sy = 0; sy < 2; ++sy) {
        float Y  = y1 + ((float)oy + ((float)sy + 0.5f) * 0.5f) * bh;
        float fy = floorf(Y);
        int   y0 = imin(imax((int)fy, 0), FH - 1);
        int   yb = imin(y0 + 1, FH - 1);
        float wy = fminf(fmaxf(Y - fy, 0.0f), 1.0f);
#pragma unroll
        for (int sx = 0; sx < 2; ++sx) {
            float Xc = x1 + ((float)ox + ((float)sx + 0.5f) * 0.5f) * bw;
            float fx = floorf(Xc);
            int   x0 = imin(imax((int)fx, 0), FW - 1);
            int   xb = imin(x0 + 1, FW - 1);
            float wx = fminf(fmaxf(Xc - fx, 0.0f), 1.0f);
            float v00 = fm[y0 * FW + x0], v01 = fm[y0 * FW + xb];
            float v10 = fm[yb * FW + x0], v11 = fm[yb * FW + xb];
            acc += (v00 * (1.0f - wx) + v01 * wx) * (1.0f - wy)
                 + (v10 * (1.0f - wx) + v11 * wx) * wy;
        }
    }
    size_t row = (size_t)b * NROI + roi;
    size_t col = (size_t)c * (POOL * POOL) + oy * POOL + ox;
    X[row * DFEAT + col] = (bf16)(acc * 0.25f);
}

// ---------------------------------------------------------------------------
// bf16 WMMA GEMM:  C(MxN) = A(MxK) * B(KxN) + bias, fp32 accumulate.
// Block tile 128x64, BK=64, 8 waves (wave32), each wave 2x2 WMMA 16x16x32
// tiles x 2 K-substeps = 8 WMMA per single barrier.
// A tile staged by GLOBAL_LOAD_ASYNC_TO_LDS_B128 (ASYNCcnt) into ping-pong
// LDS; pattern per iteration: s_wait_asynccnt 0 -> barrier -> issue next tile
// -> compute current. The single barrier orders next-buffer writes (issued
// after the barrier) against all waves' prior-iteration reads (finished
// before the barrier), so ONE barrier per K-step suffices.
// OOB A rows are clamped (row m of A only affects row m of C; rows >= M are
// never stored, so garbage there is harmless). B is staged transposed
// (sBt[n][k]) element-wise with N-bound guards so each lane's 16-K fragment
// is two contiguous 16B ds_load_b128.
// Fragment layouts per ISA 7.12.2:
//   A 16x32: lane L -> M=L%16, g=L/16; halves 0..7=K(g*8+i), 8..15=K(16+g*8+i-8)
//   B 32x16: lane L -> N=L%16, g=L/16; same K order as A
// ---------------------------------------------------------------------------
template <bool RELU, bool WF32, bool WBF16>
__global__ __launch_bounds__(256) void gemm_bf16_wmma(
    const bf16* __restrict__ A, const bf16* __restrict__ Bw,
    const float* __restrict__ bias,
    float* __restrict__ Cf, bf16* __restrict__ Cb,
    int M, int N, int K) {
    extern __shared__ char smem[];
    bf16* sA = (bf16*)smem;                  // [2][128][64]
    bf16* sB = (bf16*)(smem + 32768);        // [2][64][64]  (transposed [n][k])
    const unsigned ldsDyn = (unsigned)__builtin_amdgcn_groupstaticsize();

    const int tid  = threadIdx.x;
    const int lane = tid & 31;
    const int wv   = tid >> 5;
    const int wm   = wv & 3;   // 4 waves along M (32 rows each)
    const int wn   = wv >> 2;  // 2 waves along N (32 cols each)
    const int m0b  = blockIdx.x * 128;
    const int n0b  = blockIdx.y * 64;

    v8f acc[2][2];
#pragma unroll
    for (int i = 0; i < 2; ++i)
#pragma unroll
        for (int j = 0; j < 2; ++j) acc[i][j] = (v8f){0, 0, 0, 0, 0, 0, 0, 0};

    // A async staging: 128 rows x 128B; 4 x 16B DMA per thread
    const int arow = tid >> 1;            // 0..127
    const int acol = (tid & 1) * 32;      // element col base: 0 or 32
    int agm = m0b + arow;
    if (agm >= M) agm = M - 1;            // clamp: harmless (see header)
    const bf16* agp = A + (size_t)agm * K + acol;
    const unsigned aLds = ldsDyn + (unsigned)arow * 128u + (unsigned)acol * 2u;

    // B staging: row bk (0..63), 16 cols per thread
    const int bk  = tid >> 2;             // 0..63
    const int bn0 = (tid & 3) * 16;

    auto issueA = [&](int k0, int buf) {
        const bf16* g = agp + k0;
        unsigned l = aLds + (unsigned)buf * 16384u;
#pragma unroll
        for (int c = 0; c < 4; ++c) {
            asm volatile("global_load_async_to_lds_b128 %0, %1, off"
                         :: "v"(l + (unsigned)c * 16u), "v"(g + c * 8)
                         : "memory");
        }
    };
    auto stageB = [&](int k0, int buf) {
        int gk = k0 + bk;
        bf16* dst = sB + buf * 4096;
        if (k0 + 64 < K)
            __builtin_prefetch(Bw + (size_t)(gk + 64) * N + n0b + bn0, 0, 0);
#pragma unroll
        for (int j = 0; j < 16; ++j) {
            int gn = n0b + bn0 + j;
            bf16 v = (bf16)0.0f;
            if (gn < N) v = Bw[(size_t)gk * N + gn];
            dst[(bn0 + j) * 64 + bk] = v;
        }
    };

    issueA(0, 0);
    stageB(0, 0);

    const int g   = lane >> 4;
    const int l16 = lane & 15;
    const int T   = K >> 6;               // K / 64 (all Ks are multiples of 64)

    for (int it = 0; it < T; ++it) {
        asm volatile("s_wait_asynccnt 0x0" ::: "memory");
        __syncthreads();                  // the ONLY barrier per K-step
        if (it + 1 < T) {
            int nb = (it + 1) & 1;
            issueA((it + 1) << 6, nb);
            stageB((it + 1) << 6, nb);
        }
        const bf16* a0 = sA + (it & 1) * 8192;
        const bf16* b0 = sB + (it & 1) * 4096;
#pragma unroll
        for (int s = 0; s < 2; ++s) {
#pragma unroll
            for (int mi = 0; mi < 2; ++mi) {
                union { v16bf v; u32x4 q[2]; } af;
                int m = wm * 32 + mi * 16 + l16;
                const bf16* ap = a0 + (size_t)m * 64 + s * 32;
                af.q[0] = *(const u32x4*)(ap + g * 8);
                af.q[1] = *(const u32x4*)(ap + 16 + g * 8);
#pragma unroll
                for (int ni = 0; ni < 2; ++ni) {
                    union { v16bf v; u32x4 q[2]; } bfr;
                    int n = wn * 32 + ni * 16 + l16;
                    const bf16* bp = b0 + (size_t)n * 64 + s * 32;
                    bfr.q[0] = *(const u32x4*)(bp + g * 8);
                    bfr.q[1] = *(const u32x4*)(bp + 16 + g * 8);
                    acc[mi][ni] = __builtin_amdgcn_wmma_f32_16x16x32_bf16(
                        false, af.v, false, bfr.v, (short)0, acc[mi][ni],
                        false, false);
                }
            }
        }
    }

    // epilogue: C/D layout — VGPR r: M = r + 8*(lane/16), N = lane%16
#pragma unroll
    for (int mi = 0; mi < 2; ++mi) {
#pragma unroll
        for (int ni = 0; ni < 2; ++ni) {
            int gn = n0b + wn * 32 + ni * 16 + l16;
            if (gn >= N) continue;
            float bz = bias ? bias[gn] : 0.0f;
#pragma unroll
            for (int r = 0; r < 8; ++r) {
                int gm = m0b + wm * 32 + mi * 16 + r + 8 * g;
                if (gm >= M) continue;
                float v = acc[mi][ni][r] + bz;
                if (RELU) v = fmaxf(v, 0.0f);
                if (WF32)  Cf[(size_t)gm * N + gn] = v;
                if (WBF16) Cb[(size_t)gm * N + gn] = (bf16)v;
            }
        }
    }
}

// ---------------------------------------------------------------------------
// softmax over 81 classes, accumulate 1/3 * softmax into avg
// ---------------------------------------------------------------------------
__global__ void softmax_acc_kernel(const float* __restrict__ cls,
                                   float* __restrict__ avg, int first) {
    int row = blockIdx.x * blockDim.x + threadIdx.x;
    if (row >= MTOT) return;
    const float* p = cls + (size_t)row * NCLS;
    float mx = -1e30f;
    for (int c = 0; c < NCLS; ++c) mx = fmaxf(mx, p[c]);
    float sum = 0.0f;
    for (int c = 0; c < NCLS; ++c) sum += expf(p[c] - mx);
    float inv = (1.0f / 3.0f) / sum;
    float* a = avg + (size_t)row * NCLS;
    for (int c = 0; c < NCLS; ++c) {
        float v = expf(p[c] - mx) * inv;
        if (first) a[c] = v; else a[c] += v;
    }
}

// ---------------------------------------------------------------------------
// stage refinement: mean of decoded fg boxes, clipped
// ---------------------------------------------------------------------------
__global__ void refine_kernel(const float* __restrict__ reg,
                              const float* __restrict__ cur,
                              float* __restrict__ nxt) {
    int i = blockIdx.x * blockDim.x + threadIdx.x;
    if (i >= MTOT) return;
    const float* b = cur + (size_t)i * 4;
    float w = b[2] - b[0], h = b[3] - b[1];
    float cx = b[0] + 0.5f * w, cy = b[1] + 0.5f * h;
    float ax1 = 0, ay1 = 0, ax2 = 0, ay2 = 0;
    const float* d = reg + (size_t)i * NREG;
    for (int c = 1; c < NCLS; ++c) {
        float dx = d[c * 4 + 0] * 0.1f, dy = d[c * 4 + 1] * 0.1f;
        float dw = fminf(d[c * 4 + 2] * 0.2f, BBOX_CLIP_F);
        float dh = fminf(d[c * 4 + 3] * 0.2f, BBOX_CLIP_F);
        float pcx = dx * w + cx, pcy = dy * h + cy;
        float pw = expf(dw) * w, ph = expf(dh) * h;
        ax1 += pcx - 0.5f * pw; ay1 += pcy - 0.5f * ph;
        ax2 += pcx + 0.5f * pw; ay2 += pcy + 0.5f * ph;
    }
    const float s = 1.0f / 80.0f;
    float* o = nxt + (size_t)i * 4;
    o[0] = fminf(fmaxf(ax1 * s, 0.0f), IMGW);
    o[1] = fminf(fmaxf(ay1 * s, 0.0f), IMGH);
    o[2] = fminf(fmaxf(ax2 * s, 0.0f), IMGW);
    o[3] = fminf(fmaxf(ay2 * s, 0.0f), IMGH);
}

// final decode: pred (2000 x 81 x 4), unclipped
__global__ void decode_all_kernel(const float* __restrict__ reg,
                                  const float* __restrict__ cur,
                                  float* __restrict__ pred) {
    int i = blockIdx.x * blockDim.x + threadIdx.x;
    if (i >= MTOT * NCLS) return;
    int roi = i / NCLS, c = i % NCLS;
    const float* b = cur + (size_t)roi * 4;
    float w = b[2] - b[0], h = b[3] - b[1];
    float cx = b[0] + 0.5f * w, cy = b[1] + 0.5f * h;
    const float* d = reg + (size_t)roi * NREG + c * 4;
    float dx = d[0] * 0.1f, dy = d[1] * 0.1f;
    float dw = fminf(d[2] * 0.2f, BBOX_CLIP_F);
    float dh = fminf(d[3] * 0.2f, BBOX_CLIP_F);
    float pcx = dx * w + cx, pcy = dy * h + cy;
    float pw = expf(dw) * w, ph = expf(dh) * h;
    float* o = pred + (size_t)i * 4;
    o[0] = pcx - 0.5f * pw; o[1] = pcy - 0.5f * ph;
    o[2] = pcx + 0.5f * pw; o[3] = pcy + 0.5f * ph;
}

// ---------------------------------------------------------------------------
// candidate build: per image, 80000 (roi, fg-class) scores, padded to 2^17
// ---------------------------------------------------------------------------
__global__ void build_cand_kernel(const float* __restrict__ avg,
                                  float* __restrict__ score,
                                  int* __restrict__ idx) {
    int i = blockIdx.x * blockDim.x + threadIdx.x;  // 0 .. 2*NSORT-1
    if (i >= BIMG * NSORT) return;
    int img = i >> 17, j = i & (NSORT - 1);
    float s = -1e30f;
    if (j < NCAND) {
        int roi = j / 80, cls = 1 + (j % 80);
        float sc = avg[((size_t)img * NROI + roi) * NCLS + cls];
        s = (sc > 0.05f) ? sc : -1.0f;
    }
    score[i] = s;
    idx[i] = j;
}

// one bitonic compare-exchange step (descending), both images in one grid
__global__ void bitonic_step_kernel(float* __restrict__ score,
                                    int* __restrict__ idx, int k, int j) {
    int i = blockIdx.x * blockDim.x + threadIdx.x;
    if (i >= BIMG * NSORT) return;
    int img = i >> 17, t = i & (NSORT - 1);
    int ixj = t ^ j;
    if (ixj <= t) return;
    size_t base = (size_t)img << 17;
    float a = score[base + t], b = score[base + ixj];
    bool desc = ((t & k) == 0);
    if (desc ? (a < b) : (a > b)) {
        score[base + t] = b; score[base + ixj] = a;
        int ia = idx[base + t];
        idx[base + t] = idx[base + ixj];
        idx[base + ixj] = ia;
    }
}

// ---------------------------------------------------------------------------
// NMS + top-500 compaction. One block per image; boxes/flags in LDS (~54 KB,
// comfortably inside the 320 KB CDNA5 WGP LDS). Class-offset trick replaced
// by an exact label-equality test (offsets >= W+H+1 guarantee zero overlap).
// Output layout: fb[2*500*4] | fs[2*500] | fl[2*500]  (labels as float).
// ---------------------------------------------------------------------------
__global__ __launch_bounds__(256) void nms_kernel(
    const float* __restrict__ score, const int* __restrict__ idx,
    const float* __restrict__ pred, float* __restrict__ out) {
    const int img = blockIdx.x;
    const int tid = threadIdx.x;
    __shared__ float bx1[PRE_NMS], by1[PRE_NMS], bx2[PRE_NMS], by2[PRE_NMS];
    __shared__ float ssc[PRE_NMS];
    __shared__ int   slb[PRE_NMS];
    __shared__ unsigned char sup[PRE_NMS], keep[PRE_NMS], valid[PRE_NMS];

    size_t base = (size_t)img << 17;
    for (int t = tid; t < PRE_NMS; t += 256) {
        float s = score[base + t];
        int j = idx[base + t];
        float x1 = 0, y1 = 0, x2 = 0, y2 = 0;
        int lb = 0;
        unsigned char ok = 0;
        if (j < NCAND && s > 0.05f) {
            int roi = j / 80;
            lb = 1 + (j % 80);
            const float* p = pred + (((size_t)img * NROI + roi) * NCLS + lb) * 4;
            x1 = fminf(fmaxf(p[0], 0.0f), IMGW);
            y1 = fminf(fmaxf(p[1], 0.0f), IMGH);
            x2 = fminf(fmaxf(p[2], 0.0f), IMGW);
            y2 = fminf(fmaxf(p[3], 0.0f), IMGH);
            ok = 1;
        }
        bx1[t] = x1; by1[t] = y1; bx2[t] = x2; by2[t] = y2;
        ssc[t] = s; slb[t] = lb; valid[t] = ok;
        sup[t] = 0; keep[t] = 0;
    }
    __syncthreads();

    for (int i = 0; i < PRE_NMS; ++i) {
        if (tid == 0) keep[i] = (valid[i] && !sup[i]) ? 1 : 0;
        __syncthreads();
        if (keep[i]) {
            float ax1 = bx1[i], ay1 = by1[i], ax2 = bx2[i], ay2 = by2[i];
            float aar = (ax2 - ax1) * (ay2 - ay1);
            int alb = slb[i];
            for (int j = i + 1 + tid; j < PRE_NMS; j += 256) {
                if (slb[j] != alb) continue;  // offset classes never overlap
                float lx = fmaxf(ax1, bx1[j]), ly = fmaxf(ay1, by1[j]);
                float rx = fminf(ax2, bx2[j]), ry = fminf(ay2, by2[j]);
                float iw = fmaxf(rx - lx, 0.0f), ih = fmaxf(ry - ly, 0.0f);
                float inter = iw * ih;
                float bar = (bx2[j] - bx1[j]) * (by2[j] - by1[j]);
                float iou = inter / (aar + bar - inter + 1e-9f);
                if (iou > 0.5f) sup[j] = 1;
            }
        }
        __syncthreads();
    }

    if (tid == 0) {
        int cnt = 0;
        for (int i = 0; i < PRE_NMS && cnt < DETS; ++i) {
            if (keep[i]) {
                size_t fb = ((size_t)img * DETS + cnt) * 4;
                out[fb + 0] = bx1[i]; out[fb + 1] = by1[i];
                out[fb + 2] = bx2[i]; out[fb + 3] = by2[i];
                out[BIMG * DETS * 4 + img * DETS + cnt] = ssc[i];
                out[BIMG * DETS * 5 + img * DETS + cnt] = (float)slb[i];
                ++cnt;
            }
        }
        for (; cnt < DETS; ++cnt) {
            size_t fb = ((size_t)img * DETS + cnt) * 4;
            out[fb + 0] = 0; out[fb + 1] = 0; out[fb + 2] = 0; out[fb + 3] = 0;
            out[BIMG * DETS * 4 + img * DETS + cnt] = 0.0f;
            out[BIMG * DETS * 5 + img * DETS + cnt] = 0.0f;
        }
    }
}

// ---------------------------------------------------------------------------
extern "C" void kernel_launch(void* const* d_in, const int* in_sizes, int n_in,
                              void* d_out, int out_size, void* d_ws, size_t ws_size,
                              hipStream_t stream) {
    (void)in_sizes; (void)n_in; (void)out_size; (void)ws_size;
    const float* features  = (const float*)d_in[0];
    const float* proposals = (const float*)d_in[1];
    const float* fc1_w = (const float*)d_in[2];
    const float* fc1_b = (const float*)d_in[3];
    const float* fc2_w = (const float*)d_in[4];
    const float* fc2_b = (const float*)d_in[5];
    const float* cls_w = (const float*)d_in[6];
    const float* cls_b = (const float*)d_in[7];
    const float* reg_w = (const float*)d_in[8];
    const float* reg_b = (const float*)d_in[9];
    float* out = (float*)d_out;

    char* ws = (char*)d_ws;
    size_t off = 0;
    auto alloc = [&](size_t bytes) -> void* {
        void* p = ws + off;
        off += (bytes + 255) & ~(size_t)255;
        return p;
    };
    bf16*  X     = (bf16*)alloc((size_t)MTOT * DFEAT * 2);     // 50 MB
    bf16*  Wbuf  = (bf16*)alloc((size_t)DFEAT * HID * 2);      // 25.7 MB (reused)
    bf16*  H1    = (bf16*)alloc((size_t)MTOT * HID * 2);
    bf16*  H2    = (bf16*)alloc((size_t)MTOT * HID * 2);
    float* clsf  = (float*)alloc((size_t)MTOT * NCLS * 4);
    float* regf  = (float*)alloc((size_t)MTOT * NREG * 4);
    float* avg   = (float*)alloc((size_t)MTOT * NCLS * 4);
    float* curA  = (float*)alloc((size_t)MTOT * 4 * 4);
    float* curB  = (float*)alloc((size_t)MTOT * 4 * 4);
    float* pred  = (float*)alloc((size_t)MTOT * NCLS * 4 * 4);
    float* sc    = (float*)alloc((size_t)BIMG * NSORT * 4);
    int*   sidx  = (int*)alloc((size_t)BIMG * NSORT * 4);

    hipMemcpyAsync(curA, proposals, (size_t)MTOT * 4 * sizeof(float),
                   hipMemcpyDeviceToDevice, stream);

    float* cur = curA;
    float* nxt = curB;
    const int roiTotal = BIMG * NROI * CFEAT * POOL * POOL;
    const dim3 gFC(16, 16);   // ceil(2000/128) x (1024/64)
    const dim3 gCLS(16, 2);   // N=81  -> 2 tiles of 64
    const dim3 gREG(16, 6);   // N=324 -> 6 tiles of 64

    for (int s = 0; s < 3; ++s) {
        roi_align_kernel<<<(roiTotal + 255) / 256, 256, 0, stream>>>(
            features, cur, X, roiTotal);

        int n1 = DFEAT * HID;
        cvt_f32_bf16<<<(n1 + 255) / 256, 256, 0, stream>>>(
            fc1_w + (size_t)s * n1, Wbuf, n1);
        gemm_bf16_wmma<true, false, true><<<gFC, 256, GEMM_LDS_BYTES, stream>>>(
            X, Wbuf, fc1_b + s * HID, nullptr, H1, MTOT, HID, DFEAT);

        int n2 = HID * HID;
        cvt_f32_bf16<<<(n2 + 255) / 256, 256, 0, stream>>>(
            fc2_w + (size_t)s * n2, Wbuf, n2);
        gemm_bf16_wmma<true, false, true><<<gFC, 256, GEMM_LDS_BYTES, stream>>>(
            H1, Wbuf, fc2_b + s * HID, nullptr, H2, MTOT, HID, HID);

        int nc = HID * NCLS;
        cvt_f32_bf16<<<(nc + 255) / 256, 256, 0, stream>>>(
            cls_w + (size_t)s * nc, Wbuf, nc);
        gemm_bf16_wmma<false, true, false><<<gCLS, 256, GEMM_LDS_BYTES, stream>>>(
            H2, Wbuf, cls_b + s * NCLS, clsf, nullptr, MTOT, NCLS, HID);

        int nr = HID * NREG;
        cvt_f32_bf16<<<(nr + 255) / 256, 256, 0, stream>>>(
            reg_w + (size_t)s * nr, Wbuf, nr);
        gemm_bf16_wmma<false, true, false><<<gREG, 256, GEMM_LDS_BYTES, stream>>>(
            H2, Wbuf, reg_b + s * NREG, regf, nullptr, MTOT, NREG, HID);

        softmax_acc_kernel<<<(MTOT + 127) / 128, 128, 0, stream>>>(
            clsf, avg, s == 0 ? 1 : 0);

        if (s < 2) {
            refine_kernel<<<(MTOT + 127) / 128, 128, 0, stream>>>(regf, cur, nxt);
            float* tmp = cur; cur = nxt; nxt = tmp;
        } else {
            decode_all_kernel<<<(MTOT * NCLS + 255) / 256, 256, 0, stream>>>(
                regf, cur, pred);
        }
    }

    build_cand_kernel<<<(BIMG * NSORT) / 256, 256, 0, stream>>>(avg, sc, sidx);
    for (int k = 2; k <= NSORT; k <<= 1)
        for (int j = k >> 1; j > 0; j >>= 1)
            bitonic_step_kernel<<<(BIMG * NSORT) / 256, 256, 0, stream>>>(sc, sidx, k, j);

    nms_kernel<<<BIMG, 256, 0, stream>>>(sc, sidx, pred, out);
}
